// SA_34565896798371
// MI455X (gfx1250) — matrix operand order repaired and checked by reading, not compiled, not run on hardware
//
#include <hip/hip_runtime.h>
#include <hip/hip_bf16.h>
#include <math.h>

typedef __bf16 bf16;
typedef __attribute__((ext_vector_type(16))) __bf16 v16bf;
typedef __attribute__((ext_vector_type(8)))  float  v8f;
typedef __attribute__((ext_vector_type(4)))  int    v4i;

#define NB    32
#define NT    1024
#define DIN   768
#define NS    8
#define ND    256
#define ATT_SCALE 0.0625f     /* 256^-0.5 */
#define ATT_EPS   1e-8f
#define LN_EPS    1e-5f

/* ---------------- workspace layout (bytes) ---------------- */
static const size_t OFF_W1T  = 0;          /* w1^T  bf16 [768][768]  1179648 */
static const size_t OFF_W2T  = 1179648;    /* w2^T  bf16 [256][768]   393216 */
static const size_t OFF_KWT  = 1572864;    /* kw^T  bf16 [256][256]   131072 */
static const size_t OFF_VWT  = 1703936;    /* vw^T  bf16 [256][256]   131072 */
static const size_t OFF_WIHT = 1835008;    /* gru_wih^T f32 [256][768] 786432 */
static const size_t OFF_WHHT = 2621440;    /* gru_whh^T f32 [256][768] 786432 */
static const size_t OFF_SLOT = 3407872;    /* slots f32 [256][256]    262144 */
static const size_t OFF_Q    = 3670016;    /* q     f32 [256][256]    262144 */
static const size_t OFF_UPD  = 3932160;    /* updates f32 [256][256]  262144 */
static const size_t OFF_NEWS = 4194304;    /* GRU out f32 [256][256]  262144 */
static const size_t OFF_ATT  = 4456448;    /* attn f32 [256][1024]   1048576 */
static const size_t OFF_H    = 5505024;    /* mlp hidden f32 [256][1024] 1048576 */
static const size_t OFF_A    = 6553600;    /* 50331648: xln_bf16 -> y2_f32 -> keys_f32 */
static const size_t OFF_B    = 56885248;   /* 50331648: h1_bf16 -> feat_bf16(16MB)+values_f32(32MB) */

/* ---------------- CDNA5 async global->LDS copy ---------------- */
#if __has_builtin(__builtin_amdgcn_global_load_async_to_lds_b128)
#define HAVE_ASYNC_LDS 1
__device__ __forceinline__ void async_cp16(const bf16* g, bf16* l) {
    __builtin_amdgcn_global_load_async_to_lds_b128(
        (__attribute__((address_space(1))) v4i*)g,
        (__attribute__((address_space(3))) v4i*)l, 0, 0);
}
__device__ __forceinline__ void wait_async0() {
#if __has_builtin(__builtin_amdgcn_s_wait_asynccnt)
    __builtin_amdgcn_s_wait_asynccnt(0);
#else
    asm volatile("s_wait_asynccnt 0x0" ::: "memory");
#endif
}
#else
#define HAVE_ASYNC_LDS 0
__device__ __forceinline__ void async_cp16(const bf16* g, bf16* l) {
    *(uint4*)l = *(const uint4*)g;          /* sync fallback */
}
__device__ __forceinline__ void wait_async0() {}
#endif

/* ---------------- helpers ---------------- */
__device__ __forceinline__ float block_sum256(float v, float* red) {
    const int lane = threadIdx.x & 31, w = threadIdx.x >> 5;
#pragma unroll
    for (int o = 16; o > 0; o >>= 1) v += __shfl_down(v, o);
    if (lane == 0) red[w] = v;
    __syncthreads();
    if (w == 0) {
        float t = (lane < 8) ? red[lane] : 0.f;
#pragma unroll
        for (int o = 4; o > 0; o >>= 1) t += __shfl_down(t, o);
        if (lane == 0) red[0] = t;
    }
    __syncthreads();
    float r = red[0];
    __syncthreads();
    return r;
}

union FragU { v16bf v; uint4 u[2]; };

/* ---------------- weight prep ---------------- */
__global__ __launch_bounds__(256) void k_conv_t_bf16(const float* __restrict__ in,
                                                     bf16* __restrict__ out, int R, int C) {
    int idx = blockIdx.x * 256 + threadIdx.x;
    if (idx >= R * C) return;
    int r = idx / C, c = idx % C;
    out[(size_t)c * R + r] = (bf16)in[idx];
}

__global__ __launch_bounds__(256) void k_trans_f32(const float* __restrict__ in,
                                                   float* __restrict__ out, int R, int C) {
    int idx = blockIdx.x * 256 + threadIdx.x;
    if (idx >= R * C) return;
    int r = idx / C, c = idx % C;
    out[(size_t)c * R + r] = in[idx];
}

__global__ __launch_bounds__(256) void k_bcast_slots(const float* __restrict__ sp,
                                                     float* __restrict__ slots) {
    int i = blockIdx.x * 256 + threadIdx.x;
    slots[i] = sp[i & (NS * ND - 1)];
}

/* ---------------- LayerNorm kernels ---------------- */
__global__ __launch_bounds__(256) void k_ln_tokens(const float* __restrict__ x,
        const float* __restrict__ g, const float* __restrict__ b, bf16* __restrict__ out) {
    __shared__ float red[8];
    const int row = blockIdx.x, tid = threadIdx.x;
    const float* xr = x + (size_t)row * DIN;
    float a0 = xr[tid], a1 = xr[tid + 256], a2 = xr[tid + 512];
    float mean = block_sum256(a0 + a1 + a2, red) * (1.f / 768.f);
    float d0 = a0 - mean, d1 = a1 - mean, d2 = a2 - mean;
    float var = block_sum256(d0 * d0 + d1 * d1 + d2 * d2, red) * (1.f / 768.f);
    float rs = rsqrtf(var + LN_EPS);
    bf16* orow = out + (size_t)row * DIN;
    orow[tid]       = (bf16)(d0 * rs * g[tid]       + b[tid]);
    orow[tid + 256] = (bf16)(d1 * rs * g[tid + 256] + b[tid + 256]);
    orow[tid + 512] = (bf16)(d2 * rs * g[tid + 512] + b[tid + 512]);
}

__global__ __launch_bounds__(256) void k_ln256_bf16(const float* __restrict__ x,
        const float* __restrict__ g, const float* __restrict__ b, bf16* __restrict__ out) {
    __shared__ float red[8];
    const int row = blockIdx.x, tid = threadIdx.x;
    float a = x[(size_t)row * ND + tid];
    float mean = block_sum256(a, red) * (1.f / 256.f);
    float d = a - mean;
    float var = block_sum256(d * d, red) * (1.f / 256.f);
    out[(size_t)row * ND + tid] = (bf16)(d * rsqrtf(var + LN_EPS) * g[tid] + b[tid]);
}

/* ---------------- WMMA bf16 GEMM: C = op(A[M,K] @ Bt[N,K]^T + bias) ----------------
 * Double-buffered LDS tiles filled via GLOBAL_LOAD_ASYNC_TO_LDS_B128 (ASYNCcnt),
 * overlapping the next K-slab's copy with the current slab's ds_load + WMMA. */
#define BM 128
#define BN 128
#define BK 32
#define LDT 40               /* padded K stride in LDS (elements) */
#define TILE (BM * LDT)      /* one buffer, in elements */

template <bool RELU, bool OUTBF16>
__global__ __launch_bounds__(256) void k_gemm_bf16(const bf16* __restrict__ A,
        const bf16* __restrict__ Bt, const float* __restrict__ bias,
        float* __restrict__ Cf, bf16* __restrict__ Cb, int M, int N, int K) {
    __shared__ __attribute__((aligned(16))) bf16 sA[2 * TILE];
    __shared__ __attribute__((aligned(16))) bf16 sB[2 * TILE];
    const int tid  = threadIdx.x;
    const int lane = tid & 31;
    const int wave = tid >> 5;
    const int wm = wave >> 2;            /* 0..1: 64-row band   */
    const int wn = wave & 3;             /* 0..3: 32-col band   */
    const int bm = blockIdx.y * BM;
    const int bn = blockIdx.x * BN;
    const int half = lane >> 4;
    const int l15  = lane & 15;

    /* per-thread copy coordinates: 2 x 16B per tile per thread */
    const int r0 = tid >> 2;             /* 0..63 */
    const int sg = (tid & 3) * 8;        /* K sub-offset (elements) */
    const bf16* gA0 = A  + (size_t)(bm + r0)      * K + sg;
    const bf16* gA1 = A  + (size_t)(bm + r0 + 64) * K + sg;
    const bf16* gB0 = Bt + (size_t)(bn + r0)      * K + sg;
    const bf16* gB1 = Bt + (size_t)(bn + r0 + 64) * K + sg;
    const int lo0 = r0 * LDT + sg;
    const int lo1 = (r0 + 64) * LDT + sg;

    v8f acc[4][2];
#pragma unroll
    for (int mt = 0; mt < 4; ++mt)
#pragma unroll
        for (int nt = 0; nt < 2; ++nt)
#pragma unroll
            for (int j = 0; j < 8; ++j) acc[mt][nt][j] = 0.f;

    /* prologue: stage slab k0=0 into buffer 0 */
    async_cp16(gA0, &sA[lo0]);
    async_cp16(gA1, &sA[lo1]);
    async_cp16(gB0, &sB[lo0]);
    async_cp16(gB1, &sB[lo1]);
    wait_async0();
    __syncthreads();

    int cur = 0;
    for (int k0 = 0; k0 < K; k0 += BK) {
        const int nxt = cur ^ 1;
        const bool more = (k0 + BK) < K;
        if (more) {                       /* async-fill next buffer */
            const int nb = nxt * TILE;
            async_cp16(gA0 + k0 + BK, &sA[nb + lo0]);
            async_cp16(gA1 + k0 + BK, &sA[nb + lo1]);
            async_cp16(gB0 + k0 + BK, &sB[nb + lo0]);
            async_cp16(gB1 + k0 + BK, &sB[nb + lo1]);
            if (k0 + 2 * BK < K) {        /* L2 prefetch of slab after next */
                __builtin_prefetch(gA0 + k0 + 2 * BK, 0, 1);
                __builtin_prefetch(gB0 + k0 + 2 * BK, 0, 1);
            }
        }

        const int cb = cur * TILE;
        /* B frags: lane = column l15, 16 contiguous K at half*16 */
        FragU bf_[2];
#pragma unroll
        for (int nt = 0; nt < 2; ++nt) {
            const bf16* p = &sB[cb + (wn * 32 + nt * 16 + l15) * LDT + half * 16];
            bf_[nt].u[0] = *(const uint4*)(p);
            bf_[nt].u[1] = *(const uint4*)(p + 8);
        }
        /* A frags: lane row l15, 16 contiguous K at half*16 */
#pragma unroll
        for (int mt = 0; mt < 4; ++mt) {
            FragU af;
            const bf16* p = &sA[cb + (wm * 64 + mt * 16 + l15) * LDT + half * 16];
            af.u[0] = *(const uint4*)(p);
            af.u[1] = *(const uint4*)(p + 8);
#pragma unroll
            for (int nt = 0; nt < 2; ++nt)
                acc[mt][nt] = __builtin_amdgcn_wmma_f32_16x16x32_bf16(
                    false, af.v, false, bf_[nt].v, (short)0, acc[mt][nt],
                    false, false);
        }

        if (more) wait_async0();          /* next buffer resident */
        __syncthreads();                  /* all waves done reading cur */
        cur = nxt;
    }

    /* epilogue: D frag VGPR j -> row half*8+j, col l15 */
#pragma unroll
    for (int mt = 0; mt < 4; ++mt) {
#pragma unroll
        for (int nt = 0; nt < 2; ++nt) {
            int gc = bn + wn * 32 + nt * 16 + l15;
            float bv = bias ? bias[gc] : 0.f;
#pragma unroll
            for (int j = 0; j < 8; ++j) {
                int gr = bm + wm * 64 + mt * 16 + half * 8 + j;
                float v = acc[mt][nt][j] + bv;
                if (RELU) v = fmaxf(v, 0.f);
                if (OUTBF16) Cb[(size_t)gr * N + gc] = (bf16)v;
                else         Cf[(size_t)gr * N + gc] = v;
            }
        }
    }
}

/* ---------------- slot-attention iteration kernels (f32) ---------------- */
__global__ __launch_bounds__(256) void k_qproj(const float* __restrict__ slots,
        const float* __restrict__ g, const float* __restrict__ b,
        const float* __restrict__ qw, float* __restrict__ q) {
    __shared__ float sx[ND];
    __shared__ float red[8];
    const int row = blockIdx.x, tid = threadIdx.x;
    float a = slots[(size_t)row * ND + tid];
    float mean = block_sum256(a, red) * (1.f / 256.f);
    float d = a - mean;
    float var = block_sum256(d * d, red) * (1.f / 256.f);
    sx[tid] = d * rsqrtf(var + LN_EPS) * g[tid] + b[tid];
    __syncthreads();
    float acc = 0.f;
    for (int k = 0; k < ND; ++k) acc += sx[k] * qw[(size_t)k * ND + tid];
    q[(size_t)row * ND + tid] = acc;
}

/* one block per (b,t): 8 waves, wave s computes dot(q[b,s], key[b,t]); softmax over s */
__global__ __launch_bounds__(256) void k_dots_softmax(const float* __restrict__ q,
        const float* __restrict__ keys, float* __restrict__ attn) {
    __shared__ float d8[NS];
    const int b = blockIdx.x >> 10, t = blockIdx.x & 1023;
    const int tid = threadIdx.x, lane = tid & 31, s = tid >> 5;
    const float* qr = q + (size_t)(b * NS + s) * ND;
    const float* kr = keys + (size_t)(b * NT + t) * ND;
    float acc = 0.f;
#pragma unroll
    for (int i = 0; i < 8; ++i) { int d = lane + i * 32; acc += qr[d] * kr[d]; }
#pragma unroll
    for (int o = 16; o > 0; o >>= 1) acc += __shfl_down(acc, o);
    if (lane == 0) d8[s] = acc * ATT_SCALE;
    __syncthreads();
    if (tid < NS) {
        float mx = d8[0];
#pragma unroll
        for (int j = 1; j < NS; ++j) mx = fmaxf(mx, d8[j]);
        float den = 0.f;
#pragma unroll
        for (int j = 0; j < NS; ++j) den += expf(d8[j] - mx);
        attn[(size_t)(b * NS + tid) * NT + t] = expf(d8[tid] - mx) / den + ATT_EPS;
    }
}

__global__ __launch_bounds__(256) void k_renorm(float* __restrict__ attn) {
    __shared__ float red[8];
    const int row = blockIdx.x, tid = threadIdx.x;
    float* ar = attn + (size_t)row * NT;
    float s = 0.f;
#pragma unroll
    for (int i = 0; i < 4; ++i) s += ar[tid + i * 256];
    float inv = 1.f / block_sum256(s, red);
#pragma unroll
    for (int i = 0; i < 4; ++i) ar[tid + i * 256] *= inv;
}

__global__ __launch_bounds__(256) void k_updates(const float* __restrict__ attn,
        const float* __restrict__ values, float* __restrict__ upd) {
    __shared__ float sa[256];
    const int row = blockIdx.x, b = row >> 3, tid = threadIdx.x;
    float acc = 0.f;
    for (int t0 = 0; t0 < NT; t0 += 256) {
        sa[tid] = attn[(size_t)row * NT + t0 + tid];
        __syncthreads();
        for (int j = 0; j < 256; ++j)
            acc += sa[j] * values[(size_t)(b * NT + t0 + j) * ND + tid];
        __syncthreads();
    }
    upd[(size_t)row * ND + tid] = acc;
}

__global__ __launch_bounds__(256) void k_gru(const float* __restrict__ upd,
        const float* __restrict__ slots, const float* __restrict__ wihT,
        const float* __restrict__ whhT, const float* __restrict__ bih,
        const float* __restrict__ bhh, float* __restrict__ outs) {
    __shared__ float su[ND], sh[ND];
    const int row = blockIdx.x, d = threadIdx.x;
    su[d] = upd[(size_t)row * ND + d];
    sh[d] = slots[(size_t)row * ND + d];
    __syncthreads();
    float ir = bih[d], iz = bih[256 + d], in_ = bih[512 + d];
    float hr = bhh[d], hz = bhh[256 + d], hn = bhh[512 + d];
    for (int k = 0; k < ND; ++k) {
        float u = su[k], h = sh[k];
        const float* wi = wihT + (size_t)k * 768;
        const float* wh = whhT + (size_t)k * 768;
        ir += u * wi[d]; iz += u * wi[256 + d]; in_ += u * wi[512 + d];
        hr += h * wh[d]; hz += h * wh[256 + d]; hn  += h * wh[512 + d];
    }
    float r = 1.f / (1.f + expf(-(ir + hr)));
    float z = 1.f / (1.f + expf(-(iz + hz)));
    float n = tanhf(in_ + r * hn);
    outs[(size_t)row * ND + d] = (1.f - z) * n + z * sh[d];
}

__global__ __launch_bounds__(256) void k_mlp1(const float* __restrict__ x,
        const float* __restrict__ g, const float* __restrict__ b,
        const float* __restrict__ w1, const float* __restrict__ b1,
        float* __restrict__ h) {
    __shared__ float sx[ND];
    __shared__ float red[8];
    const int row = blockIdx.x, tid = threadIdx.x;
    float a = x[(size_t)row * ND + tid];
    float mean = block_sum256(a, red) * (1.f / 256.f);
    float d = a - mean;
    float var = block_sum256(d * d, red) * (1.f / 256.f);
    sx[tid] = d * rsqrtf(var + LN_EPS) * g[tid] + b[tid];
    __syncthreads();
#pragma unroll
    for (int c = 0; c < 4; ++c) {
        int j = tid + c * 256;
        float acc = b1[j];
        for (int k = 0; k < ND; ++k) acc += sx[k] * w1[(size_t)k * 1024 + j];
        h[(size_t)row * 1024 + j] = fmaxf(acc, 0.f);
    }
}

__global__ __launch_bounds__(256) void k_mlp2(const float* __restrict__ x,
        const float* __restrict__ h, const float* __restrict__ w2,
        const float* __restrict__ b2, float* __restrict__ slots) {
    __shared__ float sh[256];
    const int row = blockIdx.x, d = threadIdx.x;
    float acc = b2[d];
    for (int j0 = 0; j0 < 1024; j0 += 256) {
        sh[d] = h[(size_t)row * 1024 + j0 + d];
        __syncthreads();
        for (int j = 0; j < 256; ++j) acc += sh[j] * w2[(size_t)(j0 + j) * ND + d];
        __syncthreads();
    }
    slots[(size_t)row * ND + d] = x[(size_t)row * ND + d] + acc;
}

/* ---------------- launch ---------------- */
extern "C" void kernel_launch(void* const* d_in, const int* in_sizes, int n_in,
                              void* d_out, int out_size, void* d_ws, size_t ws_size,
                              hipStream_t stream) {
    const float* inputs   = (const float*)d_in[0];
    const float* slots_p  = (const float*)d_in[1];
    const float* ln_in_s  = (const float*)d_in[2];
    const float* ln_in_b  = (const float*)d_in[3];
    const float* w1_in    = (const float*)d_in[4];
    const float* b1_in    = (const float*)d_in[5];
    const float* w2_in    = (const float*)d_in[6];
    const float* b2_in    = (const float*)d_in[7];
    const float* ln_out_s = (const float*)d_in[8];
    const float* ln_out_b = (const float*)d_in[9];
    const float* qw       = (const float*)d_in[10];
    const float* kw       = (const float*)d_in[11];
    const float* vw       = (const float*)d_in[12];
    const float* norm_s   = (const float*)d_in[13];
    const float* norm_b   = (const float*)d_in[14];
    const float* gru_wih  = (const float*)d_in[15];
    const float* gru_whh  = (const float*)d_in[16];
    const float* gru_bih  = (const float*)d_in[17];
    const float* gru_bhh  = (const float*)d_in[18];
    const float* mlp_ln_s = (const float*)d_in[19];
    const float* mlp_ln_b = (const float*)d_in[20];
    const float* mlp_w1   = (const float*)d_in[21];
    const float* mlp_b1   = (const float*)d_in[22];
    const float* mlp_w2   = (const float*)d_in[23];
    const float* mlp_b2   = (const float*)d_in[24];

    char* W = (char*)d_ws;
    bf16*  w1t    = (bf16*)(W + OFF_W1T);
    bf16*  w2t    = (bf16*)(W + OFF_W2T);
    bf16*  kwt    = (bf16*)(W + OFF_KWT);
    bf16*  vwt    = (bf16*)(W + OFF_VWT);
    float* wihT   = (float*)(W + OFF_WIHT);
    float* whhT   = (float*)(W + OFF_WHHT);
    float* slots  = (float*)(W + OFF_SLOT);
    float* q      = (float*)(W + OFF_Q);
    float* upd    = (float*)(W + OFF_UPD);
    float* news   = (float*)(W + OFF_NEWS);
    float* attn   = (float*)(W + OFF_ATT);
    float* hbuf   = (float*)(W + OFF_H);
    bf16*  xln    = (bf16*)(W + OFF_A);
    float* y2     = (float*)(W + OFF_A);            /* reuse after xln dead */
    float* keys   = (float*)(W + OFF_A);            /* reuse after y2 dead  */
    bf16*  h1     = (bf16*)(W + OFF_B);
    bf16*  feat   = (bf16*)(W + OFF_B);             /* reuse after h1 dead  */
    float* values = (float*)(W + OFF_B + 16777216); /* after feat region    */

    const int M = NB * NT;                           /* 32768 token rows */

    /* weight prep */
    k_conv_t_bf16<<<(DIN * DIN + 255) / 256, 256, 0, stream>>>(w1_in, w1t, DIN, DIN);
    k_conv_t_bf16<<<(DIN * ND + 255) / 256, 256, 0, stream>>>(w2_in, w2t, DIN, ND);
    k_conv_t_bf16<<<(ND * ND + 255) / 256, 256, 0, stream>>>(kw, kwt, ND, ND);
    k_conv_t_bf16<<<(ND * ND + 255) / 256, 256, 0, stream>>>(vw, vwt, ND, ND);
    k_trans_f32<<<(768 * 256 + 255) / 256, 256, 0, stream>>>(gru_wih, wihT, 768, 256);
    k_trans_f32<<<(768 * 256 + 255) / 256, 256, 0, stream>>>(gru_whh, whhT, 768, 256);

    /* token pipeline */
    k_ln_tokens<<<M, 256, 0, stream>>>(inputs, ln_in_s, ln_in_b, xln);
    k_gemm_bf16<true, true><<<dim3(DIN / BN, M / BM), 256, 0, stream>>>(
        xln, w1t, b1_in, nullptr, h1, M, DIN, DIN);
    k_gemm_bf16<false, false><<<dim3(ND / BN, M / BM), 256, 0, stream>>>(
        h1, w2t, b2_in, y2, nullptr, M, ND, DIN);
    k_ln256_bf16<<<M, 256, 0, stream>>>(y2, ln_out_s, ln_out_b, feat);
    k_gemm_bf16<false, false><<<dim3(ND / BN, M / BM), 256, 0, stream>>>(
        feat, kwt, nullptr, keys, nullptr, M, ND, ND);
    k_gemm_bf16<false, false><<<dim3(ND / BN, M / BM), 256, 0, stream>>>(
        feat, vwt, nullptr, values, nullptr, M, ND, ND);

    /* slot attention iterations */
    k_bcast_slots<<<(NB * NS * ND) / 256, 256, 0, stream>>>(slots_p, slots);
    for (int it = 0; it < 3; ++it) {
        k_qproj<<<NB * NS, 256, 0, stream>>>(slots, norm_s, norm_b, qw, q);
        k_dots_softmax<<<NB * NT, 256, 0, stream>>>(q, keys, attn);
        k_renorm<<<NB * NS, 256, 0, stream>>>(attn);
        k_updates<<<NB * NS, 256, 0, stream>>>(attn, values, upd);
        k_gru<<<NB * NS, 256, 0, stream>>>(upd, slots, wihT, whhT, gru_bih, gru_bhh, news);
        k_mlp1<<<NB * NS, 256, 0, stream>>>(news, mlp_ln_s, mlp_ln_b, mlp_w1, mlp_b1, hbuf);
        k_mlp2<<<NB * NS, 256, 0, stream>>>(news, hbuf, mlp_w2, mlp_b2, slots);
    }

    (void)hipMemcpyAsync(d_out, slots, (size_t)NB * NS * ND * sizeof(float),
                         hipMemcpyDeviceToDevice, stream);
}
